// FusedCompactPartialUntiedLM_34368328303136
// MI455X (gfx1250) — matrix operand order repaired and checked by reading, not compiled, not run on hardware
//
#include <hip/hip_runtime.h>
#include <hip/hip_bf16.h>
#include <stdint.h>

#define DI static __device__ __forceinline__

typedef unsigned int       u32;
typedef unsigned short     u16;
typedef unsigned long long u64;

typedef float  v8f   __attribute__((ext_vector_type(8)));
typedef __bf16 v16bf __attribute__((ext_vector_type(16)));
typedef u32    u32x4 __attribute__((ext_vector_type(4)));
typedef int    i32x4 __attribute__((ext_vector_type(4)));
typedef int    i32x8 __attribute__((ext_vector_type(8)));

// ---- problem dims ----
constexpr int B_ = 2, L_ = 2048, V_ = 32000, E_ = 512, H_ = 1024, M_ = 256, P_ = 4096;
constexpr int BL = B_ * L_;        // 4096 rows
constexpr int H3 = 3 * H_;         // 3072
constexpr int FC = 4 * E_;         // 2048

// ---- bf16 helpers ----
DI u16 f2bf(float f) {
  u32 u = __float_as_uint(f);
  u32 r = u + 0x7fffu + ((u >> 16) & 1u);   // RNE
  return (u16)(r >> 16);
}
DI float bf2f(u16 h) { return __uint_as_float(((u32)h) << 16); }

DI u32 rfl(u32 v) { return (u32)__builtin_amdgcn_readfirstlane((int)v); }

// ---- Tensor Data Mover: 2D tile (tile_rows x 32 bf16) global -> LDS ----
// D# pad: interval=16 DWORDs (one 64B row), amount=4 DWORDs (16B) -> 80B LDS row pitch
// (keeps 16B alignment for ds_load_b128 and makes fragment reads bank-conflict-free)
DI void tdm_load_2d(u32 lds_byte_off, const void* gsrc, int tile_rows, int k_elems,
                    int tensor_rows) {
  u64 ga = (u64)(uintptr_t)gsrc;
  u32x4 g0;
  g0.x = rfl(1u);                                        // count=1, user mode
  g0.y = rfl(lds_byte_off);                              // lds_addr
  g0.z = rfl((u32)ga);                                   // global_addr lo
  g0.w = rfl(((u32)(ga >> 32) & 0x01FFFFFFu) | (2u << 30)); // ga hi | type=2
  u32 td0 = (u32)k_elems, td1 = (u32)tensor_rows;
  i32x8 g1;
  // data_size=1(2B) @16 | pad_enable @20 | pad_interval=3(16DW) @22 | pad_amount=3(4DW) @25
  g1[0] = (int)rfl((1u << 16) | (1u << 20) | (3u << 22) | (3u << 25));
  g1[1] = (int)rfl((td0 & 0xffffu) << 16);               // tensor_dim0[15:0]
  g1[2] = (int)rfl(((td0 >> 16) & 0xffffu) | ((td1 & 0xffffu) << 16));
  g1[3] = (int)rfl(((td1 >> 16) & 0xffffu) | (32u << 16)); // tile_dim0 = 32
  g1[4] = (int)rfl((u32)tile_rows);                      // tile_dim1 (tile_dim2=0)
  g1[5] = (int)rfl((u32)k_elems);                        // tensor_dim0_stride lo
  g1[6] = (int)rfl(0u);
  g1[7] = (int)rfl(0u);
  i32x4 z4; z4[0] = z4[1] = z4[2] = z4[3] = 0;
  i32x8 z8; z8[0] = z8[1] = z8[2] = z8[3] = z8[4] = z8[5] = z8[6] = z8[7] = 0;
  __builtin_amdgcn_tensor_load_to_lds(g0, g1, z4, z4, z8, 0);
}

// ---- generic bf16 WMMA GEMM: C[M,N] = A[M,K] * W[N,K]^T + bias ----
// mode 0: fp32 out; mode 1: relu(v)^2 -> bf16 out; mode 2: bf16 out
constexpr int TM = 64, TN = 128, TK = 32;
constexpr int APITCH = 40;   // u16 elems per LDS row (80B, matches TDM pad)

// One K=32 step for a wave's 16x64 C block: load A frag + all 4 B frags into
// distinct registers, fence the scheduler, then 4 back-to-back WMMAs
// (single s_wait_dscnt instead of one per WMMA).
DI void gemm_step(const u16* __restrict__ Ab, const u16* __restrict__ Bb,
                  int mw, int nw, int lane, v8f acc[4]) {
  union Frag { uint4 q[2]; v16bf v; };
  Frag a, b[4];
  // A 16x32 bf16 layout: lane<16 holds K0..7 & K16..23; lane>=16 K8..15 & K24..31
  const u16* ab = Ab + (mw + (lane & 15)) * APITCH + ((lane >> 4) << 3);
  a.q[0] = *(const uint4*)(ab);
  a.q[1] = *(const uint4*)(ab + 16);
#pragma unroll
  for (int s = 0; s < 4; ++s) {
    // B 32x16 bf16: lane<16 col=lane K0..7 & K16..23 ; lane>=16 col=lane-16 K8..15 & K24..31
    const u16* bb = Bb + (nw + (s << 4) + (lane & 15)) * APITCH + ((lane >> 4) << 3);
    b[s].q[0] = *(const uint4*)(bb);
    b[s].q[1] = *(const uint4*)(bb + 16);
  }
  __builtin_amdgcn_sched_barrier(0);   // keep all ds_loads before the WMMA block
#pragma unroll
  for (int s = 0; s < 4; ++s)
    acc[s] = __builtin_amdgcn_wmma_f32_16x16x32_bf16(false, a.v, false, b[s].v,
                                                     (short)0, acc[s], false, false);
}

__global__ __launch_bounds__(256) void gemm_bf16(
    const u16* __restrict__ A, const u16* __restrict__ W,
    const float* __restrict__ bias, void* __restrict__ out,
    int Mdim, int Ndim, int Kdim, int mode) {
  __shared__ __align__(16) u16 Abuf[2][TM * APITCH];
  __shared__ __align__(16) u16 Bbuf[2][TN * APITCH];
  const int n0 = blockIdx.x * TN, m0 = blockIdx.y * TM;
  const int tid = threadIdx.x, wid = tid >> 5, lane = tid & 31;
  const int mw = (wid & 3) << 4;   // wave row offset in tile
  const int nw = (wid >> 2) << 6;  // wave col offset in tile

  v8f zero = {0.f, 0.f, 0.f, 0.f, 0.f, 0.f, 0.f, 0.f};
  v8f acc[4];
#pragma unroll
  for (int s = 0; s < 4; ++s) acc[s] = zero;

  const int nk = Kdim / TK;        // even for all K used here
  if (wid == 0) {
    tdm_load_2d((u32)(uintptr_t)&Abuf[0][0], A + (size_t)m0 * Kdim, TM, Kdim, Mdim);
    tdm_load_2d((u32)(uintptr_t)&Bbuf[0][0], W + (size_t)n0 * Kdim, TN, Kdim, Ndim);
    __builtin_amdgcn_s_wait_tensorcnt(0);
  }
  __syncthreads();

  for (int kk = 0; kk < nk; kk += 2) {
    // ---- phase 0: compute buf0 (k=kk), prefetch k=kk+1 into buf1 ----
    if (wid == 0) {
      tdm_load_2d((u32)(uintptr_t)&Abuf[1][0],
                  A + (size_t)m0 * Kdim + (size_t)(kk + 1) * TK, TM, Kdim, Mdim);
      tdm_load_2d((u32)(uintptr_t)&Bbuf[1][0],
                  W + (size_t)n0 * Kdim + (size_t)(kk + 1) * TK, TN, Kdim, Ndim);
    }
    gemm_step(&Abuf[0][0], &Bbuf[0][0], mw, nw, lane, acc);
    if (wid == 0) __builtin_amdgcn_s_wait_tensorcnt(0);
    __syncthreads();
    // ---- phase 1: compute buf1 (k=kk+1), prefetch k=kk+2 into buf0 ----
    if (wid == 0 && kk + 2 < nk) {
      tdm_load_2d((u32)(uintptr_t)&Abuf[0][0],
                  A + (size_t)m0 * Kdim + (size_t)(kk + 2) * TK, TM, Kdim, Mdim);
      tdm_load_2d((u32)(uintptr_t)&Bbuf[0][0],
                  W + (size_t)n0 * Kdim + (size_t)(kk + 2) * TK, TN, Kdim, Ndim);
    }
    gemm_step(&Abuf[1][0], &Bbuf[1][0], mw, nw, lane, acc);
    if (wid == 0 && kk + 2 < nk) __builtin_amdgcn_s_wait_tensorcnt(0);
    __syncthreads();
  }

  // C layout: v8f element g -> row = base + 8*(lane>=16) + g, col = lane%16
  const int rb = m0 + mw + ((lane >> 4) << 3);
  if (mode == 0) {
    float* o = (float*)out;
#pragma unroll
    for (int s = 0; s < 4; ++s) {
      const int col = n0 + nw + (s << 4) + (lane & 15);
      const float bv = bias[col];
#pragma unroll
      for (int g = 0; g < 8; ++g)
        o[(long long)(rb + g) * Ndim + col] = acc[s][g] + bv;
    }
  } else if (mode == 1) {
    u16* o = (u16*)out;
#pragma unroll
    for (int s = 0; s < 4; ++s) {
      const int col = n0 + nw + (s << 4) + (lane & 15);
      const float bv = bias[col];
#pragma unroll
      for (int g = 0; g < 8; ++g) {
        float v = acc[s][g] + bv;
        v = v > 0.f ? v * v : 0.f;      // relu then square
        o[(long long)(rb + g) * Ndim + col] = f2bf(v);
      }
    }
  } else {
    u16* o = (u16*)out;
#pragma unroll
    for (int s = 0; s < 4; ++s) {
      const int col = n0 + nw + (s << 4) + (lane & 15);
      const float bv = bias[col];
#pragma unroll
      for (int g = 0; g < 8; ++g)
        o[(long long)(rb + g) * Ndim + col] = f2bf(acc[s][g] + bv);
    }
  }
}

// ---- small utility kernels ----
__global__ void init_kernel(int* t2p, u32* bar) {
  int i = blockIdx.x * 256 + threadIdx.x;
  if (i < V_) t2p[i] = -1;
  if (i == 0) bar[0] = 0;  // barrier counter (generation bar[1] is self-consistent)
}
__global__ void t2p_scatter(int* t2p, const int* __restrict__ untied) {
  int p = blockIdx.x * 256 + threadIdx.x;
  if (p < P_) t2p[untied[p]] = p;
}
__global__ void slots_kernel(int* slots, const int* __restrict__ ids,
                             const int* __restrict__ t2p) {
  int r = blockIdx.x * 256 + threadIdx.x;
  if (r < BL) slots[r] = t2p[ids[r]];
}
__global__ void cast_kernel(u16* dst, const float* __restrict__ src, int n) {
  int i = blockIdx.x * 256 + threadIdx.x;
  if (i < n) dst[i] = f2bf(src[i]);
}
__global__ void gather_kernel(u16* xb, const float* __restrict__ emb,
                              const int* __restrict__ ids) {
  const int row = blockIdx.x;
  const int tok = ids[row];
  for (int c = threadIdx.x; c < E_; c += 256)
    xb[(size_t)row * E_ + c] = f2bf(emb[(size_t)tok * E_ + c]);
}

// ---- device-wide barrier (sense via generation counter) ----
DI void grid_barrier(u32* cnt, u32* gen, u32 nwg) {
  __syncthreads();
  if (threadIdx.x == 0) {
    __threadfence();
    volatile u32* vg = (volatile u32*)gen;
    u32 g = *vg;
    if (atomicAdd(cnt, 1u) == nwg - 1u) {
      *(volatile u32*)cnt = 0u;
      __threadfence();
      atomicAdd(gen, 1u);
    } else {
      while (*vg == g) __builtin_amdgcn_s_sleep(2);
    }
    __threadfence();
  }
  __syncthreads();
}

// ---- GRU: 32 WGs, each owns 32 h-columns (all 3 gates local; W_hh slice in LDS bf16) ----
constexpr int GRU_WG = 32, JW = 32;
constexpr int GRU_LDS = 3 * JW * H_ * 2 + 2 * H_ * 4 + 192 * 4;  // 205,568 B

__global__ __launch_bounds__(256) void gru_kernel(
    const float* __restrict__ w_hh, const float* __restrict__ b_hh,
    const float* __restrict__ xp, float* __restrict__ states,
    float* __restrict__ hbuf2, u32* __restrict__ bar) {
  extern __shared__ __align__(16) char sm[];
  u16* wl = (u16*)sm;                           // [3*JW][H_] bf16
  float* hb = (float*)(sm + 3 * JW * H_ * 2);   // [2][H_]
  float* pd = hb + 2 * H_;                      // [192]
  const int g = blockIdx.x, tid = threadIdx.x;

  for (int idx = tid; idx < 3 * JW * H_; idx += 256) {
    int gate = idx / (JW * H_), rem = idx % (JW * H_);
    int jl = rem / H_, kc = rem % H_;
    wl[idx] = f2bf(w_hh[(size_t)(gate * H_ + g * JW + jl) * H_ + kc]);
  }
  if (tid < 2 * JW) {                 // zero h ping-buffer 0 (own slice)
    int b = tid / JW, jl = tid % JW;
    hbuf2[b * H_ + g * JW + jl] = 0.f;
  }
  grid_barrier(bar, bar + 1, GRU_WG);

  for (int t = 0; t < L_; ++t) {
    const int rdbuf = t & 1;
    const float* hg = hbuf2 + rdbuf * (2 * H_);
    for (int idx = tid; idx < 2 * H_; idx += 256) hb[idx] = hg[idx];
    __syncthreads();
    if (tid < 192) {                  // 3 gates x 2 batch x 32 cols dots, K=1024
      const int gate = tid / 64, r = tid % 64, b = r / JW, jl = r % JW;
      const u16* wr = wl + (gate * JW + jl) * H_;
      const float* hr = hb + b * H_;
      float acc = 0.f;
      for (int kc = 0; kc < H_; kc += 8) {
        uint4 w8 = *(const uint4*)(wr + kc);
        float4 h0 = *(const float4*)(hr + kc);
        float4 h1 = *(const float4*)(hr + kc + 4);
        acc += bf2f((u16)(w8.x & 0xffffu)) * h0.x;
        acc += bf2f((u16)(w8.x >> 16)) * h0.y;
        acc += bf2f((u16)(w8.y & 0xffffu)) * h0.z;
        acc += bf2f((u16)(w8.y >> 16)) * h0.w;
        acc += bf2f((u16)(w8.z & 0xffffu)) * h1.x;
        acc += bf2f((u16)(w8.z >> 16)) * h1.y;
        acc += bf2f((u16)(w8.w & 0xffffu)) * h1.z;
        acc += bf2f((u16)(w8.w >> 16)) * h1.w;
      }
      pd[tid] = acc;
    }
    __syncthreads();
    if (tid < 2 * JW) {               // fuse gates, update h
      const int b = tid / JW, jl = tid % JW, j = g * JW + jl;
      const size_t row = (size_t)b * L_ + t;
      const float xr = xp[row * H3 + j];
      const float xz = xp[row * H3 + H_ + j];
      const float xn = xp[row * H3 + 2 * H_ + j];
      const float hrv = pd[tid] + b_hh[j];
      const float hzv = pd[64 + tid] + b_hh[H_ + j];
      const float hnv = pd[128 + tid] + b_hh[2 * H_ + j];
      const float rg = 1.f / (1.f + __expf(-(xr + hrv)));
      const float zg = 1.f / (1.f + __expf(-(xz + hzv)));
      const float ng = tanhf(xn + rg * hnv);
      const float hp = hb[b * H_ + j];
      const float hn2 = (1.f - zg) * ng + zg * hp;
      hbuf2[(rdbuf ^ 1) * (2 * H_) + b * H_ + j] = hn2;
      states[row * H_ + j] = hn2;
    }
    grid_barrier(bar, bar + 1, GRU_WG);
  }
}

// ---- gate = sigmoid(states @ wg^T + bg), one wave per row ----
__global__ __launch_bounds__(256) void gate_kernel(
    const float* __restrict__ states, const float* __restrict__ wg,
    const float* __restrict__ bg, float* __restrict__ gate) {
  const int wid = threadIdx.x >> 5, lane = threadIdx.x & 31;
  const int row = blockIdx.x * 8 + wid;
  const float* sr = states + (size_t)row * H_;
  float acc = 0.f;
  for (int c = lane * 4; c < H_; c += 128) {
    float4 s4 = *(const float4*)(sr + c);
    float4 w4 = *(const float4*)(wg + c);
    acc += s4.x * w4.x + s4.y * w4.y + s4.z * w4.z + s4.w * w4.w;
  }
  for (int off = 16; off; off >>= 1) acc += __shfl_xor(acc, off, 32);
  if (lane == 0) gate[row] = 1.f / (1.f + __expf(-(acc + bg[0])));
}

// ---- causal attention row + softmax + gated scatter into P slots (LDS resident) ----
__global__ __launch_bounds__(256) void attn_kernel(
    const float* __restrict__ q, const float* __restrict__ k,
    const float* __restrict__ gate, const int* __restrict__ slots,
    const float* __restrict__ mscale, float* __restrict__ memp) {
  __shared__ __align__(16) float qs[M_];
  __shared__ float sc[L_];
  __shared__ float mp[P_];
  __shared__ float red[9];
  const int row = blockIdx.x;                 // b*L + i
  const int b = row / L_, i = row % L_;
  const int tid = threadIdx.x, wid = tid >> 5, lane = tid & 31;
  qs[tid] = q[(size_t)row * M_ + tid];
  for (int p = tid; p < P_; p += 256) mp[p] = 0.f;
  __syncthreads();
  float lmax = -3.0e38f;
  for (int j = tid; j < i; j += 256) {
    const float* kr = k + ((size_t)b * L_ + j) * M_;
    float d = 0.f;
    for (int c = 0; c < M_; c += 4) {
      float4 q4 = *(const float4*)(qs + c);
      float4 k4 = *(const float4*)(kr + c);
      d += q4.x * k4.x + q4.y * k4.y + q4.z * k4.z + q4.w * k4.w;
    }
    d *= 0.0625f;  // 1/sqrt(256)
    sc[j] = d;
    lmax = fmaxf(lmax, d);
  }
  for (int off = 16; off; off >>= 1) lmax = fmaxf(lmax, __shfl_xor(lmax, off, 32));
  if (lane == 0) red[wid] = lmax;
  __syncthreads();
  if (tid == 0) {
    float m = red[0];
    for (int w = 1; w < 8; ++w) m = fmaxf(m, red[w]);
    red[8] = m;
  }
  __syncthreads();
  const float mx = red[8];
  float lsum = 0.f;
  for (int j = tid; j < i; j += 256) {
    float e = __expf(sc[j] - mx);
    sc[j] = e;
    lsum += e;
  }
  for (int off = 16; off; off >>= 1) lsum += __shfl_xor(lsum, off, 32);
  if (lane == 0) red[wid] = lsum;
  __syncthreads();
  if (tid == 0) {
    float s = 0.f;
    for (int w = 0; w < 8; ++w) s += red[w];
    red[8] = s;
  }
  __syncthreads();
  const float denom = fmaxf(red[8], 1e-6f);
  const float scale = gate[row] * mscale[0] / denom;
  for (int j = tid; j < i; j += 256) {
    const int s = slots[b * L_ + j];
    if (s >= 0) atomicAdd(&mp[s], sc[j] * scale);  // repeated tokens -> LDS atomic
  }
  __syncthreads();
  for (int p = tid; p < P_; p += 256)
    memp[(size_t)row * P_ + p] = mp[p];
}

// ---- final: out[row, untied[p]] += total_partial + mem_partial ----
__global__ void final_scatter(float* __restrict__ out, const float* __restrict__ tp,
                              const float* __restrict__ memp,
                              const int* __restrict__ untied) {
  const int row = blockIdx.x;
  for (int p = threadIdx.x; p < P_; p += 256) {
    const int col = untied[p];
    out[(size_t)row * V_ + col] +=
        tp[(size_t)row * P_ + p] + memp[(size_t)row * P_ + p];
  }
}

extern "C" void kernel_launch(void* const* d_in, const int* in_sizes, int n_in,
                              void* d_out, int out_size, void* d_ws, size_t ws_size,
                              hipStream_t stream) {
  (void)in_sizes; (void)n_in; (void)out_size; (void)ws_size;
  const int*   ids    = (const int*)d_in[0];
  const int*   untied = (const int*)d_in[1];
  const float* emb    = (const float*)d_in[2];
  const float* w_ih   = (const float*)d_in[3];
  const float* w_hh   = (const float*)d_in[4];
  const float* b_ih   = (const float*)d_in[5];
  const float* b_hh   = (const float*)d_in[6];
  const float* wq     = (const float*)d_in[7];
  const float* bq     = (const float*)d_in[8];
  const float* wk     = (const float*)d_in[9];
  const float* bk     = (const float*)d_in[10];
  const float* wg     = (const float*)d_in[11];
  const float* bg     = (const float*)d_in[12];
  const float* w_fc   = (const float*)d_in[13];
  const float* b_fc   = (const float*)d_in[14];
  const float* w_proj = (const float*)d_in[15];
  const float* b_proj = (const float*)d_in[16];
  const float* obias  = (const float*)d_in[17];
  const float* w_part = (const float*)d_in[18];
  const float* b_part = (const float*)d_in[19];
  const float* mscale = (const float*)d_in[20];
  float* out = (float*)d_out;

  char* ws = (char*)d_ws;
  size_t off = 0;
  auto alloc = [&](size_t bytes) -> char* {
    char* p = ws + off;
    off += (bytes + 255) & ~(size_t)255;
    return p;
  };
  u16*   emb_bf    = (u16*)alloc((size_t)V_ * E_ * 2);
  u16*   wih_bf    = (u16*)alloc((size_t)H3 * E_ * 2);
  u16*   wfc_bf    = (u16*)alloc((size_t)FC * H_ * 2);
  u16*   wproj_bf  = (u16*)alloc((size_t)E_ * FC * 2);
  u16*   wq_bf     = (u16*)alloc((size_t)M_ * H_ * 2);
  u16*   wk_bf     = (u16*)alloc((size_t)M_ * H_ * 2);
  u16*   wpart_bf  = (u16*)alloc((size_t)P_ * E_ * 2);
  u16*   x_bf      = (u16*)alloc((size_t)BL * E_ * 2);
  float* xp        = (float*)alloc((size_t)BL * H3 * 4);
  float* states    = (float*)alloc((size_t)BL * H_ * 4);
  u16*   states_bf = (u16*)alloc((size_t)BL * H_ * 2);
  u16*   head_bf   = (u16*)alloc((size_t)BL * FC * 2);
  u16*   bfeat_bf  = (u16*)alloc((size_t)BL * E_ * 2);
  float* qb        = (float*)alloc((size_t)BL * M_ * 4);
  float* kb        = (float*)alloc((size_t)BL * M_ * 4);
  float* gateb     = (float*)alloc((size_t)BL * 4);
  float* memp      = (float*)alloc((size_t)BL * P_ * 4);
  float* tp        = (float*)alloc((size_t)BL * P_ * 4);
  int*   t2p       = (int*)alloc((size_t)V_ * 4);
  int*   slots     = (int*)alloc((size_t)BL * 4);
  float* hbuf2     = (float*)alloc((size_t)2 * 2 * H_ * 4);
  u32*   bar       = (u32*)alloc(256);

  init_kernel<<<(V_ + 255) / 256, 256, 0, stream>>>(t2p, bar);
  t2p_scatter<<<(P_ + 255) / 256, 256, 0, stream>>>(t2p, untied);
  slots_kernel<<<(BL + 255) / 256, 256, 0, stream>>>(slots, ids, t2p);

  auto cast = [&](u16* d, const float* s, size_t n) {
    cast_kernel<<<(unsigned)((n + 255) / 256), 256, 0, stream>>>(d, s, (int)n);
  };
  cast(emb_bf, emb, (size_t)V_ * E_);
  cast(wih_bf, w_ih, (size_t)H3 * E_);
  cast(wfc_bf, w_fc, (size_t)FC * H_);
  cast(wproj_bf, w_proj, (size_t)E_ * FC);
  cast(wq_bf, wq, (size_t)M_ * H_);
  cast(wk_bf, wk, (size_t)M_ * H_);
  cast(wpart_bf, w_part, (size_t)P_ * E_);
  gather_kernel<<<BL, 256, 0, stream>>>(x_bf, emb, ids);

  dim3 blk(256);
  // xp = x @ W_ih^T + b_ih   (fp32 out, feeds GRU)
  gemm_bf16<<<dim3(H3 / TN, BL / TM), blk, 0, stream>>>(x_bf, wih_bf, b_ih, xp, BL, H3, E_, 0);
  // sequential GRU
  gru_kernel<<<GRU_WG, 256, GRU_LDS, stream>>>(w_hh, b_hh, xp, states, hbuf2, bar);
  cast(states_bf, states, (size_t)BL * H_);
  // head = relu(states @ W_fc^T + b_fc)^2  -> bf16
  gemm_bf16<<<dim3(FC / TN, BL / TM), blk, 0, stream>>>(states_bf, wfc_bf, b_fc, head_bf, BL, FC, H_, 1);
  // base_feat = head @ W_proj^T + b_proj -> bf16
  gemm_bf16<<<dim3(E_ / TN, BL / TM), blk, 0, stream>>>(head_bf, wproj_bf, b_proj, bfeat_bf, BL, E_, FC, 2);
  // q, k
  gemm_bf16<<<dim3(M_ / TN, BL / TM), blk, 0, stream>>>(states_bf, wq_bf, bq, qb, BL, M_, H_, 0);
  gemm_bf16<<<dim3(M_ / TN, BL / TM), blk, 0, stream>>>(states_bf, wk_bf, bk, kb, BL, M_, H_, 0);
  gate_kernel<<<BL / 8, 256, 0, stream>>>(states, wg, bg, gateb);
  // base_logits = base_feat @ emb^T + output_bias  -> d_out (dominant GEMM)
  gemm_bf16<<<dim3(V_ / TN, BL / TM), blk, 0, stream>>>(bfeat_bf, emb_bf, obias, out, BL, V_, E_, 0);
  // attention -> mem_partial
  attn_kernel<<<BL, 256, 0, stream>>>(qb, kb, gateb, slots, mscale, memp);
  // total_partial (matmul part) = base_feat @ w_partial^T + b_partial
  gemm_bf16<<<dim3(P_ / TN, BL / TM), blk, 0, stream>>>(bfeat_bf, wpart_bf, b_part, tp, BL, P_, E_, 0);
  // out[:, untied] += total_partial + mem_partial
  final_scatter<<<BL, 256, 0, stream>>>(out, tp, memp, untied);
}
